// UAActor_35570919145519
// MI455X (gfx1250) — matrix operand order repaired and checked by reading, not compiled, not run on hardware
//
#include <hip/hip_runtime.h>
#include <math.h>

typedef __attribute__((ext_vector_type(16))) __bf16          v16bf;
typedef __attribute__((ext_vector_type(16))) unsigned short  v16us;
typedef __attribute__((ext_vector_type(8)))  unsigned short  v8us;
typedef __attribute__((ext_vector_type(8)))  float           v8f;

#define DEVFN static __device__ __forceinline__

constexpr int Bsz = 32768, Ssz = 512, Asz = 8, H1 = 400, H2 = 300, NMC = 3;
constexpr int H1P = 416;   // GEMM2 K padded to 13*32
constexpr int H2R = 304;   // GEMM2 N padded to 19*16 (w2 storage rows)
constexpr int NT2 = 19;    // GEMM2 column tiles
constexpr int H2S = 320;   // LDS stride for layer-2 acts = GEMM3 K padded to 10*32

// ---- dynamic LDS layout ----
constexpr int UNI_E   = 2 * 16 * H1P;                 // 13312 >= 16*Ssz (8192)
constexpr int OFF_LM1 = 0;
constexpr int OFF_LV1 = OFF_LM1 + 16 * H1P;
constexpr int OFF_UNI = OFF_LV1 + 16 * H1P;
constexpr int OFF_H2M = OFF_UNI + UNI_E;
constexpr int OFF_H2V = OFF_H2M + 16 * H2S;
constexpr int OFF_F32 = OFF_H2V + 16 * H2S;           // u16 elements so far (even)
constexpr size_t SMEM_BYTES = (size_t)OFF_F32 * 2 + 128 * 4;   // + svb[128] f32

// workspace layout (bytes)
constexpr size_t WS_MNMX = 0;                                   // 2 x u32
constexpr size_t WS_W1BF = 256;                                 // 400*512 u16
constexpr size_t WS_W2BF = WS_W1BF + (size_t)H1 * Ssz * 2;      // 304*416 u16
constexpr size_t WS_W2SQ = WS_W2BF + (size_t)H2R * H1P * 2;
constexpr size_t WS_W1SQ = WS_W2SQ + (size_t)H2R * H1P * 2;     // 400 f32
constexpr size_t WS_W3BF = WS_W1SQ + 1600;                      // 16*320 u16 (B-layout, padded)
constexpr size_t WS_W3SQ = WS_W3BF + (size_t)16 * H2S * 2;

DEVFN unsigned short f2bf(float f) {
  return __builtin_bit_cast(unsigned short, (__bf16)f);
}
DEVFN float bf2f(unsigned short u) {
  return (float)__builtin_bit_cast(__bf16, u);
}

DEVFN unsigned enc_ord(float f) {            // order-preserving f32 -> u32
  unsigned u = __float_as_uint(f);
  return (u & 0x80000000u) ? ~u : (u | 0x80000000u);
}
DEVFN float dec_ord(unsigned e) {
  unsigned u = (e & 0x80000000u) ? (e ^ 0x80000000u) : ~e;
  return __uint_as_float(u);
}

// branch-free erf (Abramowitz-Stegun 7.1.26, |err|<1.5e-7): 1 rcp + 1 exp + FMAs
DEVFN float fast_erf(float z) {
  float az = fabsf(z);
  float t  = __frcp_rn(__builtin_fmaf(0.3275911f, az, 1.0f));
  float p  = t * (0.254829592f +
             t * (-0.284496736f +
             t * (1.421413741f +
             t * (-1.453152027f + t * 1.061405429f))));
  float r = 1.0f - p * __expf(-az * az);
  return copysignf(r, z);
}

DEVFN void adf_relu(float m, float v, float& om, float& ov) {
  float vv = v + 1e-20f;
  float rs = __frsqrt_rn(vv);                 // 1/sqrt(v)
  float sd = vv * rs;                         // sqrt(v)
  float r  = m * rs;
  float cdf = 0.5f * (1.0f + fast_erf(r * 0.7071067811865476f));
  float pdf = 0.3989422804014327f * __expf(-0.5f * r * r);
  om = m * cdf + sd * pdf;
  ov = (m * m + v) * cdf + m * sd * pdf - om * om;
}

// WMMA A-operand from LDS: two contiguous 16B chunks (k 16*half+0..7 / 16*half+8..15)
DEVFN v16bf load_A(const unsigned short* base, int off_lo, int off_hi) {
  v8us lo = *reinterpret_cast<const v8us*>(base + off_lo);
  v8us hi = *reinterpret_cast<const v8us*>(base + off_hi);
  v16us c = __builtin_shufflevector(lo, hi, 0, 1, 2, 3, 4, 5, 6, 7,
                                    8, 9, 10, 11, 12, 13, 14, 15);
  return __builtin_bit_cast(v16bf, c);
}

// ---------------------------------------------------------------- prep kernels
__global__ void k_init(unsigned* mnmx) { mnmx[0] = 0xFFFFFFFFu; mnmx[1] = 0u; }

__global__ void k_minmax(const float* __restrict__ x, unsigned* __restrict__ mnmx) {
  __shared__ unsigned smn[256], smx[256];
  unsigned lmn = 0xFFFFFFFFu, lmx = 0u;
  const size_t n = (size_t)Bsz * Ssz;
  for (size_t i = (size_t)blockIdx.x * 256 + threadIdx.x; i < n;
       i += (size_t)gridDim.x * 256) {
    unsigned e = enc_ord(x[i]);
    lmn = (e < lmn) ? e : lmn;
    lmx = (e > lmx) ? e : lmx;
  }
  smn[threadIdx.x] = lmn; smx[threadIdx.x] = lmx;
  __syncthreads();
  for (int s = 128; s > 0; s >>= 1) {
    if ((int)threadIdx.x < s) {
      unsigned a = smn[threadIdx.x + s]; if (a < smn[threadIdx.x]) smn[threadIdx.x] = a;
      unsigned b = smx[threadIdx.x + s]; if (b > smx[threadIdx.x]) smx[threadIdx.x] = b;
    }
    __syncthreads();
  }
  if (threadIdx.x == 0) { atomicMin(&mnmx[0], smn[0]); atomicMax(&mnmx[1], smx[0]); }
}

__global__ void k_prep_w1(const float* __restrict__ W1, unsigned short* __restrict__ w1bf) {
  int i = blockIdx.x * 256 + threadIdx.x;
  if (i < H1 * Ssz) w1bf[i] = f2bf(W1[i]);
}

__global__ void k_prep_w2(const float* __restrict__ W2, unsigned short* __restrict__ w2bf,
                          unsigned short* __restrict__ w2sq) {
  int i = blockIdx.x * 256 + threadIdx.x;
  if (i >= H2R * H1P) return;
  int r = i / H1P, c = i % H1P;
  float w = (r < H2 && c < H1) ? W2[r * H1 + c] : 0.0f;   // zero padding
  w2bf[i] = f2bf(w);
  w2sq[i] = f2bf(w * w);
}

__global__ void k_prep_w3(const float* __restrict__ W3, unsigned short* __restrict__ w3bf,
                          unsigned short* __restrict__ w3sq) {
  int i = blockIdx.x * 256 + threadIdx.x;
  if (i >= 16 * H2S) return;
  int a = i / H2S, k = i % H2S;
  float w = (a < Asz && k < H2) ? W3[a * H2 + k] : 0.0f;  // zero padding (N 8->16, K 300->320)
  w3bf[i] = f2bf(w);
  w3sq[i] = f2bf(w * w);
}

__global__ void k_w1sq(const float* __restrict__ W1, float* __restrict__ w1sq) {
  int h = blockIdx.x * 256 + threadIdx.x;
  if (h >= H1) return;
  float acc = 0.f;
  for (int s = 0; s < Ssz; ++s) { float w = W1[h * Ssz + s]; acc += w * w; }
  w1sq[h] = acc;
}

// ---------------------------------------------------------------- fused pipeline
__global__ __launch_bounds__(256) void k_fused(
    const float* __restrict__ x,  const float* __restrict__ b1,
    const float* __restrict__ b2, const float* __restrict__ b3,
    const float* __restrict__ mask1, const float* __restrict__ mask2,
    const unsigned short* __restrict__ w1bf,
    const unsigned short* __restrict__ w2bf,
    const unsigned short* __restrict__ w2sq,
    const unsigned short* __restrict__ w3bf,
    const unsigned short* __restrict__ w3sq,
    const float* __restrict__ w1sq,
    const unsigned* __restrict__ mnmx,
    float* __restrict__ out) {
  extern __shared__ char smem[];
  unsigned short* lm1  = (unsigned short*)smem + OFF_LM1;
  unsigned short* lv1  = (unsigned short*)smem + OFF_LV1;
  unsigned short* lmn  = (unsigned short*)smem + OFF_UNI;        // stage 0/1
  unsigned short* lam  = (unsigned short*)smem + OFF_UNI;        // stage 2 (aliases lmn)
  unsigned short* lav  = lam + 16 * H1P;
  unsigned short* lh2m = (unsigned short*)smem + OFF_H2M;
  unsigned short* lh2v = (unsigned short*)smem + OFF_H2V;
  float* svb = (float*)((unsigned short*)smem + OFF_F32);        // 128 f32

  const int tid  = threadIdx.x;
  const int lane = tid & 31;
  const int wv   = tid >> 5;     // wave 0..7
  const int half = lane >> 4;    // 0/1 : K-half for A/B operands
  const int ln   = lane & 15;    // row (A) / column (B,C)
  const size_t row0 = (size_t)blockIdx.x * 16;

  const float xmin = dec_ord(mnmx[0]);
  const float xmax = dec_ord(mnmx[1]);
  const float inv_rng = 1.0f / (xmax - xmin);

  // ---- Stage 0: zero lh2 K-pad cols (304..319); normalize x-tile -> bf16 LDS ----
  { int r = tid >> 4, c = H2R + (tid & 15);
    lh2m[r * H2S + c] = 0; lh2v[r * H2S + c] = 0; }
  for (int i = tid; i < 16 * Ssz; i += 256) {
    int r = i >> 9, c = i & (Ssz - 1);
    float f = (x[(row0 + (size_t)r) * Ssz + c] - xmin) * inv_rng + 0.1f;
    lmn[i] = f2bf(f);
  }
  __syncthreads();

  // ---- Stage 1: m1 = mn @ W1^T + b1 ; v1 = 2.0001*||W1_h||^2 ; ADF ReLU ----
  for (int ct = wv; ct < H1 / 16; ct += 8) {            // 25 column tiles
    v8f acc = (v8f)(0.0f);
    const int h = ct * 16 + ln;
    for (int k0 = 0; k0 < Ssz; k0 += 32) {
      v16bf Am = load_A(lmn, ln * Ssz + k0 + 16 * half,
                             ln * Ssz + k0 + 16 * half + 8);
      v16bf Bm = *reinterpret_cast<const v16bf*>(w1bf + h * Ssz + k0 + 16 * half);
      acc = __builtin_amdgcn_wmma_f32_16x16x32_bf16(false, Am, false, Bm,
                                                    (short)0, acc, false, false);
    }
    const float bias = b1[h];
    const float vcol = w1sq[h] * 2.0001f;               // (NOISE_VAR + 2)*||W1_h||^2
    #pragma unroll
    for (int j = 0; j < 8; ++j) {
      int r = j + 8 * half;
      float om, ov; adf_relu(acc[j] + bias, vcol, om, ov);
      lm1[r * H1P + h] = f2bf(om);
      lv1[r * H1P + h] = f2bf(ov);
    }
  }
  __syncthreads();     // lm1/lv1 ready; lmn dead -> region reused as lam/lav

  float bias3 = 0.0f;
  if (wv == 0 && ln < Asz) bias3 = b3[ln];
  v8f s1v = (v8f)(0.0f), s2v = (v8f)(0.0f), svv = (v8f)(0.0f);

  for (int n = 0; n < NMC; ++n) {
    // ---- Stage 2a: apply mask1 once per block -> masked activations (bf16) ----
    for (int i = tid; i < 16 * H1P; i += 256) {
      int r = i / H1P, k = i - r * H1P;
      if (k < H1) {
        float mk = mask1[((size_t)n * Bsz + row0 + (size_t)r) * H1 + k];
        lam[i] = f2bf(bf2f(lm1[i]) * mk);
        lav[i] = f2bf(bf2f(lv1[i]) * (mk * mk));
      } else {
        lam[i] = 0; lav[i] = 0;           // K padding
      }
    }
    // prefetch the mask tiles consumed after the upcoming GEMM (warm L2/WGP$)
    {
      const char* p2 = (const char*)(mask2 + ((size_t)n * Bsz + row0) * H2);
      if (tid < 75) __builtin_prefetch(p2 + tid * 256, 0, 0);      // 16*300*4 B
      if (n + 1 < NMC) {
        const char* p1 = (const char*)(mask1 + ((size_t)(n + 1) * Bsz + row0) * H1);
        if (tid >= 128 && tid < 228) __builtin_prefetch(p1 + (tid - 128) * 256, 0, 0);
      }
    }
    __syncthreads();

    // ---- Stage 2: dual GEMM (mean vs W2, variance vs W2^2) ----
    for (int ct = wv; ct < NT2; ct += 8) {              // 19 column tiles
      v8f accm = (v8f)(0.0f), accv = (v8f)(0.0f);
      const int h2 = ct * 16 + ln;
      for (int k0 = 0; k0 < H1P; k0 += 32) {            // 13 K-steps
        v16bf Am = load_A(lam, ln * H1P + k0 + 16 * half,
                               ln * H1P + k0 + 16 * half + 8);
        v16bf Av = load_A(lav, ln * H1P + k0 + 16 * half,
                               ln * H1P + k0 + 16 * half + 8);
        v16bf Bm = *reinterpret_cast<const v16bf*>(w2bf + h2 * H1P + k0 + 16 * half);
        v16bf Bv = *reinterpret_cast<const v16bf*>(w2sq + h2 * H1P + k0 + 16 * half);
        accm = __builtin_amdgcn_wmma_f32_16x16x32_bf16(false, Am, false, Bm,
                                                       (short)0, accm, false, false);
        accv = __builtin_amdgcn_wmma_f32_16x16x32_bf16(false, Av, false, Bv,
                                                       (short)0, accv, false, false);
      }
      const float bias2 = (h2 < H2) ? b2[h2] : 0.0f;
      #pragma unroll
      for (int j = 0; j < 8; ++j) {
        int r = j + 8 * half;
        float om, ov; adf_relu(accm[j] + bias2, accv[j], om, ov);
        lh2m[r * H2S + h2] = f2bf(om);
        lh2v[r * H2S + h2] = f2bf(ov);
      }
    }
    __syncthreads();

    // ---- mask2 applied in LDS ----
    for (int i = tid; i < 16 * H2; i += 256) {
      int r = i / H2, h = i - r * H2;
      float mk = mask2[((size_t)n * Bsz + row0 + (size_t)r) * H2 + h];
      lh2m[r * H2S + h] = f2bf(bf2f(lh2m[r * H2S + h]) * mk);
      lh2v[r * H2S + h] = f2bf(bf2f(lh2v[r * H2S + h]) * mk * mk);
    }
    __syncthreads();

    // ---- Stage 3: [16x320]x[320x16] WMMA (N padded 8->16) ----
    if (wv == 0) {                         // mean path: lh2m x W3^T
      v8f acc = (v8f)(0.0f);
      for (int k0 = 0; k0 < H2S; k0 += 32) {
        v16bf Am = load_A(lh2m, ln * H2S + k0 + 16 * half,
                                ln * H2S + k0 + 16 * half + 8);
        v16bf Bm = *reinterpret_cast<const v16bf*>(w3bf + ln * H2S + k0 + 16 * half);
        acc = __builtin_amdgcn_wmma_f32_16x16x32_bf16(false, Am, false, Bm,
                                                      (short)0, acc, false, false);
      }
      #pragma unroll
      for (int j = 0; j < 8; ++j) {
        float mv = acc[j] + bias3;
        s1v[j] += mv; s2v[j] += mv * mv;
      }
    } else if (wv == 1) {                  // variance path: lh2v x (W3^2)^T
      v8f acc = (v8f)(0.0f);
      for (int k0 = 0; k0 < H2S; k0 += 32) {
        v16bf Av = load_A(lh2v, ln * H2S + k0 + 16 * half,
                                ln * H2S + k0 + 16 * half + 8);
        v16bf Bv = *reinterpret_cast<const v16bf*>(w3sq + ln * H2S + k0 + 16 * half);
        acc = __builtin_amdgcn_wmma_f32_16x16x32_bf16(false, Av, false, Bv,
                                                      (short)0, acc, false, false);
      }
      #pragma unroll
      for (int j = 0; j < 8; ++j) svv[j] += acc[j];
    }
    __syncthreads();   // protect lh2 / lam before next sample overwrites
  }

  // ---- combine MC statistics, write outputs ----
  if (wv == 1 && ln < Asz) {
    #pragma unroll
    for (int j = 0; j < 8; ++j) svb[(j + 8 * half) * Asz + ln] = svv[j];
  }
  __syncthreads();
  if (wv == 0 && ln < Asz) {
    #pragma unroll
    for (int j = 0; j < 8; ++j) {
      int r = j + 8 * half;
      float mean = s1v[j] * (1.0f / 3.0f);
      float var  = svb[r * Asz + ln] * (1.0f / 3.0f)
                 + s2v[j] * (1.0f / 3.0f) - mean * mean;
      size_t o = (row0 + (size_t)r) * Asz + ln;
      out[o] = mean;
      out[(size_t)Bsz * Asz + o] = __expf(var * 2.302585092994046f);  // 10^var
    }
  }
}

// ---------------------------------------------------------------- launch
extern "C" void kernel_launch(void* const* d_in, const int* in_sizes, int n_in,
                              void* d_out, int out_size, void* d_ws, size_t ws_size,
                              hipStream_t stream) {
  (void)in_sizes; (void)n_in; (void)out_size; (void)ws_size;
  const float* x     = (const float*)d_in[0];
  // d_in[1] = x_noise : unused by the reference computation
  const float* W1    = (const float*)d_in[2];
  const float* b1    = (const float*)d_in[3];
  const float* W2    = (const float*)d_in[4];
  const float* b2    = (const float*)d_in[5];
  const float* W3    = (const float*)d_in[6];
  const float* b3    = (const float*)d_in[7];
  const float* mask1 = (const float*)d_in[8];
  const float* mask2 = (const float*)d_in[9];
  float* out = (float*)d_out;

  char* ws = (char*)d_ws;
  unsigned*       mnmx = (unsigned*)(ws + WS_MNMX);
  unsigned short* w1bf = (unsigned short*)(ws + WS_W1BF);
  unsigned short* w2bf = (unsigned short*)(ws + WS_W2BF);
  unsigned short* w2sq = (unsigned short*)(ws + WS_W2SQ);
  float*          w1sq = (float*)(ws + WS_W1SQ);
  unsigned short* w3bf = (unsigned short*)(ws + WS_W3BF);
  unsigned short* w3sq = (unsigned short*)(ws + WS_W3SQ);

  k_init<<<1, 1, 0, stream>>>(mnmx);
  k_minmax<<<1024, 256, 0, stream>>>(x, mnmx);
  k_prep_w1<<<(H1 * Ssz + 255) / 256, 256, 0, stream>>>(W1, w1bf);
  k_prep_w2<<<(H2R * H1P + 255) / 256, 256, 0, stream>>>(W2, w2bf, w2sq);
  k_prep_w3<<<(16 * H2S + 255) / 256, 256, 0, stream>>>(W3, w3bf, w3sq);
  k_w1sq<<<(H1 + 255) / 256, 256, 0, stream>>>(W1, w1sq);
  k_fused<<<Bsz / 16, 256, (unsigned)SMEM_BYTES, stream>>>(
      x, b1, b2, b3, mask1, mask2, w1bf, w2bf, w2sq, w3bf, w3sq, w1sq, mnmx, out);
}
